// GATDynamicsNetwork_17806934409717
// MI455X (gfx1250) — matrix operand order repaired and checked by reading, not compiled, not run on hardware
//
#include <hip/hip_runtime.h>

// ---------------------------------------------------------------------------
// Types for CDNA5 WMMA (wave32)
// ---------------------------------------------------------------------------
typedef __attribute__((ext_vector_type(16))) _Float16 v16h;
typedef __attribute__((ext_vector_type(8)))  _Float16 v8h;
typedef __attribute__((ext_vector_type(8)))  float    v8f;

// Fixed problem sizes
#define BB     8192
#define NNODE  16
#define NHEAD  4
#define DHEAD  64
#define NEDGE  48

// Grid-graph edges (same enumeration as the reference _make_edges)
__constant__ int ESRC[NEDGE] = {
  0,1,0,4, 1,2,1,5, 2,3,2,6, 3,7,
  4,5,4,8, 5,6,5,9, 6,7,6,10, 7,11,
  8,9,8,12, 9,10,9,13, 10,11,10,14, 11,15,
  12,13, 13,14, 14,15};
__constant__ int EDST[NEDGE] = {
  1,0,4,0, 2,1,5,1, 3,2,6,2, 7,3,
  5,4,8,4, 6,5,9,5, 7,6,10,6, 11,7,
  9,8,12,8, 10,9,13,9, 11,10,14,10, 15,11,
  13,12, 14,13, 15,14};

// ---------------------------------------------------------------------------
// Weight convert: fp32 [Kr,Nr] row-major -> f16 transposed [Np][Kp], zero pad
// ---------------------------------------------------------------------------
__global__ void conv_w_kernel(const float* __restrict__ W, _Float16* __restrict__ Wt,
                              int Kr, int Nr, int Kp, int Np) {
  int idx = blockIdx.x * blockDim.x + threadIdx.x;
  if (idx >= Kp * Np) return;
  int k = idx % Kp, n = idx / Kp;
  float v = (k < Kr && n < Nr) ? W[k * Nr + n] : 0.0f;
  Wt[(long)n * Kp + k] = (_Float16)v;
}

__global__ void pad_bias_kernel(const float* __restrict__ b, float* __restrict__ out,
                                int Nr, int Np) {
  int i = blockIdx.x * blockDim.x + threadIdx.x;
  if (i < Np) out[i] = (i < Nr) ? b[i] : 0.0f;
}

// ---------------------------------------------------------------------------
// X0 = concat(state[B,256], emb[action]) -> f16 [B,320]
// ---------------------------------------------------------------------------
__global__ void prep_x0_kernel(const float* __restrict__ state, const int* __restrict__ action,
                               const float* __restrict__ emb, _Float16* __restrict__ X0) {
  long i = (long)blockIdx.x * blockDim.x + threadIdx.x;
  if (i >= (long)BB * 320) return;
  int b = (int)(i / 320), c = (int)(i % 320);
  float v = (c < 256) ? state[(long)b * 256 + c] : emb[action[b] * 64 + (c - 256)];
  X0[i] = (_Float16)v;
}

// grid[B,256] f16 -> node features [B*16, 32] f16 (cols 16..31 zero, K padded)
// x[b,n,c] = grid[b, c*16 + n]
__global__ void permute_nodes_kernel(const _Float16* __restrict__ grid,
                                     _Float16* __restrict__ xf) {
  long i = (long)blockIdx.x * blockDim.x + threadIdx.x;
  if (i >= (long)BB * NNODE * 32) return;
  int c = (int)(i & 31);
  long bn = i >> 5;
  int n = (int)(bn & 15);
  long b = bn >> 4;
  xf[i] = (c < 16) ? grid[b * 256 + c * 16 + n] : (_Float16)0.0f;
}

// ---------------------------------------------------------------------------
// Generic WMMA GEMM: C[M,Np] = A[M,K](f16) @ Wt[Np,K](f16,transposed) + bias
// One wave computes a 16 x (NT*16) tile, software-pipelined: fragments for
// k-step i+1 are loaded into a second live buffer while the NT
// v_wmma_f32_16x16x32_f16 for k-step i issue, so the waits before the WMMAs
// cover loads issued a full iteration earlier (latency overlapped) instead of
// a full s_wait_loadcnt 0 drain. A-operand traffic shrinks NT-fold vs
// one-tile-per-wave. Tile index is wave-uniform (EXEC all-ones).
// ---------------------------------------------------------------------------
template <int NT>
struct Frag {
  v8h a0, a1;
  v8h b0[NT], b1[NT];
};

template <int NT>
__device__ __forceinline__ void load_frag(Frag<NT>& f, const _Float16* ap,
                                          const _Float16* const* bp, int k, int hsel) {
  // A fragment (16-bit A 16x32 ISA layout): lane holds row (lane&15),
  // halves K = k + hsel*8 .. +7  and  K = k + 16 + hsel*8 .. +7
  f.a0 = *(const v8h*)(ap + k + hsel * 8);
  f.a1 = *(const v8h*)(ap + k + 16 + hsel * 8);
  // B fragments: lane holds col (lane&15), K = k + hsel*8 .. +7 and K = k + 16 + hsel*8 .. +7
  #pragma unroll
  for (int j = 0; j < NT; ++j) {
    f.b0[j] = *(const v8h*)(bp[j] + k + hsel * 8);
    f.b1[j] = *(const v8h*)(bp[j] + k + 16 + hsel * 8);
  }
}

template <int NT>
__device__ __forceinline__ void mma_frag(const Frag<NT>& f, v8f* acc) {
  v16h av;
  #pragma unroll
  for (int i = 0; i < 8; ++i) { av[i] = f.a0[i]; av[i + 8] = f.a1[i]; }
  #pragma unroll
  for (int j = 0; j < NT; ++j) {
    v16h bv;
    #pragma unroll
    for (int i = 0; i < 8; ++i) { bv[i] = f.b0[j][i]; bv[i + 8] = f.b1[j][i]; }
    acc[j] = __builtin_amdgcn_wmma_f32_16x16x32_f16(false, av, false, bv,
                                                    (short)0, acc[j], false, false);
  }
}

template <int NT>
__global__ void gemm_wmma_kernel(const _Float16* __restrict__ A,
                                 const _Float16* __restrict__ Bt,
                                 const float* __restrict__ bias,
                                 float* __restrict__ Cf,
                                 _Float16* __restrict__ Ch,
                                 int M, int Np, int K, int relu) {
  const int lane = threadIdx.x & 31;
  const int wid  = threadIdx.x >> 5;
  const int ntn  = Np >> 4;        // total 16-wide N tiles
  const int ntb  = ntn / NT;       // N tile-blocks per row of tiles
  long tile = (long)blockIdx.x * (blockDim.x >> 5) + wid;
  long total = ((long)M >> 4) * ntb;
  if (tile >= total) return;       // wave-uniform exit
  const int  tnb = (int)(tile % ntb);
  const long tm  = tile / ntb;

  const int hsel = lane >> 4;      // which K half-group this lane holds
  const int arow = (int)(tm * 16) + (lane & 15);
  const _Float16* ap = A + (long)arow * K;

  const _Float16* bp[NT];
  int bcol[NT];
  #pragma unroll
  for (int j = 0; j < NT; ++j) {
    bcol[j] = (tnb * NT + j) * 16 + (lane & 15);
    bp[j]   = Bt + (long)bcol[j] * K;
  }

  v8f acc[NT];
  #pragma unroll
  for (int j = 0; j < NT; ++j) acc[j] = (v8f){};

  // Software pipeline: prologue load, then {load k | mma k-32}, epilogue mma.
  Frag<NT> cur, nxt;
  load_frag<NT>(cur, ap, bp, 0, hsel);
  for (int k = 32; k < K; k += 32) {
    load_frag<NT>(nxt, ap, bp, k, hsel);
    mma_frag<NT>(cur, acc);
    cur = nxt;
  }
  mma_frag<NT>(cur, acc);

  // C/D layout: lane (l&15)=col, VGPR v -> row (l>>4)*8 + v
  const int rbase = (int)(tm * 16) + hsel * 8;
  #pragma unroll
  for (int j = 0; j < NT; ++j) {
    const float bv = bias ? bias[bcol[j]] : 0.0f;
    #pragma unroll
    for (int v = 0; v < 8; ++v) {
      float x = acc[j][v] + bv;
      if (relu) x = fmaxf(x, 0.0f);
      long off = (long)(rbase + v) * Np + bcol[j];
      if (Cf) Cf[off] = x;
      if (Ch) Ch[off] = (_Float16)x;
    }
  }
}

// ---------------------------------------------------------------------------
// LayerNorm + ReLU, one wave per row (wave32 shuffle reduction)
// ---------------------------------------------------------------------------
__global__ void ln_relu_kernel(const float* __restrict__ X,
                               const float* __restrict__ g, const float* __restrict__ be,
                               _Float16* __restrict__ Oh, float* __restrict__ Of,
                               int M, int F, int SIn, int SOut) {
  const int lane = threadIdx.x & 31;
  const int wid  = threadIdx.x >> 5;
  long row = (long)blockIdx.x * (blockDim.x >> 5) + wid;
  if (row >= M) return;
  const float* x = X + row * SIn;
  float s = 0.0f, s2 = 0.0f;
  for (int i = lane; i < F; i += 32) { float v = x[i]; s += v; s2 += v * v; }
  for (int m = 16; m; m >>= 1) { s += __shfl_xor(s, m, 32); s2 += __shfl_xor(s2, m, 32); }
  float mean = s / F;
  float var  = s2 / F - mean * mean;
  float rs   = rsqrtf(var + 1e-5f);
  for (int i = lane; i < F; i += 32) {
    float y = (x[i] - mean) * rs * g[i] + be[i];
    y = fmaxf(y, 0.0f);
    if (Oh) Oh[row * SOut + i] = (_Float16)y;
    if (Of) Of[row * SOut + i] = y;
  }
}

// ---------------------------------------------------------------------------
// GAT attention. One wave per sample; 16x256 node block staged in LDS.
// ---------------------------------------------------------------------------
#define AW 4   // waves (samples) per block

__device__ __forceinline__ void gat_alpha_common(
    const _Float16* __restrict__ Hlin, const float* __restrict__ asrc,
    const float* __restrict__ adst, int b, int lane, int wid,
    _Float16 (*sh)[NNODE * 256], float (*s_al)[2][64], float (*s_esc)[192],
    float (*s_alpha)[192]) {
  // 1. stage h (16 nodes x 256) into LDS: 4096 halves = 512 uint4
  const uint4* src = (const uint4*)(Hlin + (long)b * NNODE * 256);
  uint4* dst = (uint4*)sh[wid];
  for (int i = lane; i < 512; i += 32) dst[i] = src[i];
  __syncthreads();
  // 2. attention logits al_s / al_d per (node, head)
  for (int t = lane; t < 64; t += 32) {
    int n = t >> 2, hd = t & 3;
    float as = 0.0f, ad = 0.0f;
    for (int d = 0; d < DHEAD; ++d) {
      float v = (float)sh[wid][n * 256 + hd * DHEAD + d];
      as += v * asrc[hd * DHEAD + d];
      ad += v * adst[hd * DHEAD + d];
    }
    s_al[wid][0][t] = as; s_al[wid][1][t] = ad;
  }
  __syncthreads();
  // 3. per-edge scores + leaky relu
  for (int t = lane; t < 192; t += 32) {
    int e = t >> 2, hd = t & 3;
    float sc = s_al[wid][0][ESRC[e] * 4 + hd] + s_al[wid][1][EDST[e] * 4 + hd];
    s_esc[wid][t] = sc > 0.0f ? sc : 0.2f * sc;
  }
  __syncthreads();
  // 4. softmax over incoming edges per (dst node, head) -> alpha per edge
  for (int t = lane; t < 64; t += 32) {
    int n = t >> 2, hd = t & 3;
    float m = -1e30f;
    for (int e = 0; e < NEDGE; ++e)
      if (EDST[e] == n) m = fmaxf(m, s_esc[wid][e * 4 + hd]);
    float den = 0.0f;
    for (int e = 0; e < NEDGE; ++e)
      if (EDST[e] == n) den += __expf(s_esc[wid][e * 4 + hd] - m);
    for (int e = 0; e < NEDGE; ++e)
      if (EDST[e] == n) s_alpha[wid][e * 4 + hd] = __expf(s_esc[wid][e * 4 + hd] - m) / den;
  }
  __syncthreads();
}

// gat0: full concat output [B*16, 256], + bias + relu
__global__ void gat_attn0_kernel(const _Float16* __restrict__ Hlin,
                                 const float* __restrict__ asrc,
                                 const float* __restrict__ adst,
                                 const float* __restrict__ bias,
                                 _Float16* __restrict__ Out) {
  __shared__ __align__(16) _Float16 sh[AW][NNODE * 256];
  __shared__ float s_al[AW][2][64];
  __shared__ float s_esc[AW][192];
  __shared__ float s_alpha[AW][192];
  const int lane = threadIdx.x & 31, wid = threadIdx.x >> 5;
  const int b = blockIdx.x * AW + wid;
  gat_alpha_common(Hlin, asrc, adst, b, lane, wid, sh, s_al, s_esc, s_alpha);
  // 5. aggregate: per (node, head): out[d] = sum_in alpha * h[src]
  for (int q = lane; q < 64; q += 32) {
    int n = q >> 2, hd = q & 3;
    int cnt = 0, se[4]; float al[4];
    for (int e = 0; e < NEDGE; ++e)
      if (EDST[e] == n) { se[cnt] = ESRC[e]; al[cnt] = s_alpha[wid][e * 4 + hd]; ++cnt; }
    for (int d = 0; d < DHEAD; ++d) {
      float acc = 0.0f;
      for (int i = 0; i < cnt; ++i)
        acc += al[i] * (float)sh[wid][se[i] * 256 + hd * DHEAD + d];
      float y = acc + bias[hd * DHEAD + d];
      Out[((long)b * NNODE + n) * 256 + hd * DHEAD + d] = (_Float16)fmaxf(y, 0.0f);
    }
  }
}

// gat1: head-mean + node-mean pooling fused -> pooled[B,64] (+ bias)
__global__ void gat_attn1_kernel(const _Float16* __restrict__ Hlin,
                                 const float* __restrict__ asrc,
                                 const float* __restrict__ adst,
                                 const float* __restrict__ bias,
                                 _Float16* __restrict__ Pooled) {
  __shared__ __align__(16) _Float16 sh[AW][NNODE * 256];
  __shared__ float s_al[AW][2][64];
  __shared__ float s_esc[AW][192];
  __shared__ float s_alpha[AW][192];
  const int lane = threadIdx.x & 31, wid = threadIdx.x >> 5;
  const int b = blockIdx.x * AW + wid;
  gat_alpha_common(Hlin, asrc, adst, b, lane, wid, sh, s_al, s_esc, s_alpha);
  // pooled[d] = bias[d] + (1/(N*H)) * sum_{e,hd} alpha[e,hd] * h[src_e, hd, d]
  for (int d = lane; d < DHEAD; d += 32) {
    float acc = 0.0f;
    for (int hd = 0; hd < NHEAD; ++hd)
      for (int e = 0; e < NEDGE; ++e)
        acc += s_alpha[wid][e * 4 + hd] * (float)sh[wid][ESRC[e] * 256 + hd * DHEAD + d];
    Pooled[(long)b * DHEAD + d] = (_Float16)(bias[d] + acc * (1.0f / 64.0f));
  }
}

// ---------------------------------------------------------------------------
// Host-side orchestration
// ---------------------------------------------------------------------------
static inline int ceil_div(long a, long b) { return (int)((a + b - 1) / b); }

extern "C" void kernel_launch(void* const* d_in, const int* in_sizes, int n_in,
                              void* d_out, int out_size, void* d_ws, size_t ws_size,
                              hipStream_t stream) {
  (void)in_sizes; (void)n_in; (void)out_size; (void)ws_size;
  const float* state  = (const float*)d_in[0];
  const int*   action = (const int*)  d_in[1];
  const float* emb    = (const float*)d_in[2];

  char* ws = (char*)d_ws;
  size_t off = 0;
  auto alloc = [&](size_t bytes) -> char* {
    char* p = ws + off;
    off = (off + bytes + 255) & ~(size_t)255;
    return p;
  };

  // Converted f16 transposed weights [Np][Kp]
  struct WSpec { int src; int Kr, Nr, Kp, Np; _Float16* dst; };
  WSpec wspec[12] = {
    {3,  320, 128, 320, 128, nullptr},  // fusion0
    {7,  128, 128, 128, 128, nullptr},  // fusion1
    {11, 128, 256, 128, 256, nullptr},  // fusion2
    {15, 256, 256, 256, 256, nullptr},  // proj
    {17, 16,  64,  32,  64,  nullptr},  // in-lin (K padded 16->32)
    {19, 64,  256, 64,  256, nullptr},  // gat0 W
    {23, 256, 256, 256, 256, nullptr},  // gat1 W
    {27, 64,  128, 64,  128, nullptr},  // out_mlp0
    {31, 128, 256, 128, 256, nullptr},  // out_mlp1
    {35, 256, 64,  256, 64,  nullptr},  // reward0
    {39, 64,  64,  64,  64,  nullptr},  // reward1
    {43, 64,  601, 64,  608, nullptr},  // reward2 (N padded 601->608)
  };
  for (int i = 0; i < 12; ++i)
    wspec[i].dst = (_Float16*)alloc((size_t)wspec[i].Kp * wspec[i].Np * 2);

  float*     br2   = (float*)    alloc(608 * 4);                        // padded reward2 bias
  _Float16*  X0    = (_Float16*) alloc((size_t)BB * 320 * 2);
  float*     T32   = (float*)    alloc((size_t)BB * 608 * 4);           // f32 GEMM scratch
  _Float16*  actA  = (_Float16*) alloc((size_t)BB * 320 * 2);
  _Float16*  actB  = (_Float16*) alloc((size_t)BB * 320 * 2);
  _Float16*  grid16= (_Float16*) alloc((size_t)BB * 256 * 2);
  _Float16*  nodef = (_Float16*) alloc((size_t)BB * NNODE * 32 * 2);
  _Float16*  h0    = (_Float16*) alloc((size_t)BB * NNODE * 64 * 2);
  _Float16*  BIG0  = (_Float16*) alloc((size_t)BB * NNODE * 256 * 2);
  _Float16*  BIG1  = (_Float16*) alloc((size_t)BB * NNODE * 256 * 2);
  _Float16*  pooled= (_Float16*) alloc((size_t)BB * 64 * 2);

  // --- convert weights ---
  for (int i = 0; i < 12; ++i) {
    int n = wspec[i].Kp * wspec[i].Np;
    conv_w_kernel<<<ceil_div(n, 256), 256, 0, stream>>>(
        (const float*)d_in[wspec[i].src], wspec[i].dst,
        wspec[i].Kr, wspec[i].Nr, wspec[i].Kp, wspec[i].Np);
  }
  pad_bias_kernel<<<3, 256, 0, stream>>>((const float*)d_in[44], br2, 601, 608);

  // --- input prep ---
  prep_x0_kernel<<<ceil_div((long)BB * 320, 256), 256, 0, stream>>>(state, action, emb, X0);

  auto gemm = [&](const _Float16* A, const _Float16* Bt, const float* bias,
                  float* Cf, _Float16* Ch, int M, int Np, int K, int relu) {
    int ntn = Np >> 4;
    if ((ntn & 3) == 0) {
      long tiles = ((long)M >> 4) * (ntn >> 2);
      gemm_wmma_kernel<4><<<ceil_div(tiles, 8), 256, 0, stream>>>(A, Bt, bias, Cf, Ch, M, Np, K, relu);
    } else if ((ntn & 1) == 0) {
      long tiles = ((long)M >> 4) * (ntn >> 1);
      gemm_wmma_kernel<2><<<ceil_div(tiles, 8), 256, 0, stream>>>(A, Bt, bias, Cf, Ch, M, Np, K, relu);
    } else {
      long tiles = ((long)M >> 4) * ntn;
      gemm_wmma_kernel<1><<<ceil_div(tiles, 8), 256, 0, stream>>>(A, Bt, bias, Cf, Ch, M, Np, K, relu);
    }
  };
  auto ln = [&](const float* X, const float* g, const float* be,
                _Float16* Oh, float* Of, int M, int F, int SIn, int SOut) {
    ln_relu_kernel<<<ceil_div(M, 8), 256, 0, stream>>>(X, g, be, Oh, Of, M, F, SIn, SOut);
  };

  // --- fusion MLP: 320 -> 128 -> 128 -> 256 (Linear -> LN -> ReLU each) ---
  gemm(X0,   wspec[0].dst, (const float*)d_in[4],  T32, nullptr, BB, 128, 320, 0);
  ln(T32, (const float*)d_in[5],  (const float*)d_in[6],  actA, nullptr, BB, 128, 128, 128);
  gemm(actA, wspec[1].dst, (const float*)d_in[8],  T32, nullptr, BB, 128, 128, 0);
  ln(T32, (const float*)d_in[9],  (const float*)d_in[10], actB, nullptr, BB, 128, 128, 128);
  gemm(actB, wspec[2].dst, (const float*)d_in[12], T32, nullptr, BB, 256, 128, 0);
  ln(T32, (const float*)d_in[13], (const float*)d_in[14], actA, nullptr, BB, 256, 256, 256);

  // --- proj + permute to node features ---
  gemm(actA, wspec[3].dst, (const float*)d_in[16], nullptr, grid16, BB, 256, 256, 0);
  permute_nodes_kernel<<<ceil_div((long)BB * NNODE * 32, 256), 256, 0, stream>>>(grid16, nodef);

  // --- in-lin (16->64, relu) ---
  gemm(nodef, wspec[4].dst, (const float*)d_in[18], nullptr, h0, BB * NNODE, 64, 32, 1);

  // --- GAT layer 0 ---
  gemm(h0, wspec[5].dst, nullptr, nullptr, BIG0, BB * NNODE, 256, 64, 0);
  gat_attn0_kernel<<<BB / AW, AW * 32, 0, stream>>>(
      BIG0, (const float*)d_in[20], (const float*)d_in[21], (const float*)d_in[22], BIG1);

  // --- GAT layer 1 (+ fused head-mean / node-mean pooling) ---
  gemm(BIG1, wspec[6].dst, nullptr, nullptr, BIG0, BB * NNODE, 256, 256, 0);
  gat_attn1_kernel<<<BB / AW, AW * 32, 0, stream>>>(
      BIG0, (const float*)d_in[24], (const float*)d_in[25], (const float*)d_in[26], pooled);

  // --- out_mlp: 64 -> 128 -> 256, next_state written to d_out (f32) ---
  float* next_state = (float*)d_out;
  float* reward_out = (float*)d_out + (size_t)BB * 256;
  gemm(pooled, wspec[7].dst, (const float*)d_in[28], T32, nullptr, BB, 128, 64, 0);
  ln(T32, (const float*)d_in[29], (const float*)d_in[30], actA, nullptr, BB, 128, 128, 128);
  gemm(actA, wspec[8].dst, (const float*)d_in[32], T32, nullptr, BB, 256, 128, 0);
  ln(T32, (const float*)d_in[33], (const float*)d_in[34], actB, next_state, BB, 256, 256, 256);

  // --- reward head: 256 -> 64 -> 64 -> 601 ---
  gemm(actB, wspec[9].dst, (const float*)d_in[36], T32, nullptr, BB, 64, 256, 0);
  ln(T32, (const float*)d_in[37], (const float*)d_in[38], actA, nullptr, BB, 64, 64, 64);
  gemm(actA, wspec[10].dst, (const float*)d_in[40], T32, nullptr, BB, 64, 64, 0);
  ln(T32, (const float*)d_in[41], (const float*)d_in[42], actB, nullptr, BB, 64, 64, 64);
  gemm(actB, wspec[11].dst, br2, T32, nullptr, BB, 608, 64, 0);
  ln(T32, (const float*)d_in[45], (const float*)d_in[46], nullptr, reward_out, BB, 601, 608, 601);
}
